// SetAbstractionLayer_13219909337188
// MI455X (gfx1250) — matrix operand order repaired and checked by reading, not compiled, not run on hardware
//
#include <hip/hip_runtime.h>

// ---------------------------------------------------------------------------
// PointNet++ SetAbstraction (FPS -> ball query -> PointConv MLP -> max pool)
// Sizes fixed by the reference: B=16, N=4096, C=64, ratio=0.25 -> S=1024,
// radius=0.2, K=32, MLP 67->128->256.
// ---------------------------------------------------------------------------

#define BB 16
#define NN 4096
#define CC 64
#define SS 1024
#define KK 32
#define R2 0.04f

typedef __attribute__((ext_vector_type(16))) _Float16 v16h;
typedef __attribute__((ext_vector_type(8)))  float    v8f;

// ---------------------------------------------------------------------------
// Kernel 1: farthest point sampling. One block per cloud, 1024 threads,
// 4 points per thread. Deterministic: start at point 0, argmax ties -> lowest
// index (matches jnp.argmax).
// ---------------------------------------------------------------------------
__global__ __launch_bounds__(1024) void fps_kernel(const float* __restrict__ pos,
                                                   int* __restrict__ samp,
                                                   float* __restrict__ pos_s) {
    const int b   = blockIdx.x;
    const int tid = threadIdx.x;
    const float* pb = pos + (size_t)b * NN * 3;

    __shared__ float sd[1024];
    __shared__ int   si[1024];

    const float x0 = pb[0], y0 = pb[1], z0 = pb[2];

    float px[4], py[4], pz[4], dmin[4];
#pragma unroll
    for (int t = 0; t < 4; ++t) {
        const int p = tid * 4 + t;
        px[t] = pb[p * 3 + 0];
        py[t] = pb[p * 3 + 1];
        pz[t] = pb[p * 3 + 2];
        const float dx = px[t] - x0, dy = py[t] - y0, dz = pz[t] - z0;
        dmin[t] = dx * dx + dy * dy + dz * dz;
    }

    if (tid == 0) {
        samp[b * SS + 0] = 0;
        pos_s[(size_t)(b * SS + 0) * 3 + 0] = x0;
        pos_s[(size_t)(b * SS + 0) * 3 + 1] = y0;
        pos_s[(size_t)(b * SS + 0) * 3 + 2] = z0;
    }

    for (int i = 1; i < SS; ++i) {
        // local argmax over this thread's 4 points (first max wins)
        float bd = dmin[0];
        int   bi = tid * 4;
#pragma unroll
        for (int t = 1; t < 4; ++t) {
            if (dmin[t] > bd) { bd = dmin[t]; bi = tid * 4 + t; }
        }
        sd[tid] = bd;
        si[tid] = bi;
        __syncthreads();
        // tree argmax; on tie keep lower index (lower thread id holds lower
        // point indices)
        for (int off = 512; off > 0; off >>= 1) {
            if (tid < off) {
                const float od = sd[tid + off];
                if (od > sd[tid]) { sd[tid] = od; si[tid] = si[tid + off]; }
            }
            __syncthreads();
        }
        const int nxt = si[0];
        __syncthreads();   // everyone read si[0] before next iter overwrites

        const float nx = pb[nxt * 3 + 0];
        const float ny = pb[nxt * 3 + 1];
        const float nz = pb[nxt * 3 + 2];
        if (tid == 0) {
            samp[b * SS + i] = nxt;
            pos_s[(size_t)(b * SS + i) * 3 + 0] = nx;
            pos_s[(size_t)(b * SS + i) * 3 + 1] = ny;
            pos_s[(size_t)(b * SS + i) * 3 + 2] = nz;
        }
#pragma unroll
        for (int t = 0; t < 4; ++t) {
            const float dx = px[t] - nx, dy = py[t] - ny, dz = pz[t] - nz;
            const float d2 = dx * dx + dy * dy + dz * dz;
            dmin[t] = fminf(dmin[t], d2);
        }
    }
}

// ---------------------------------------------------------------------------
// Kernel 2: ball query / K-nearest-within-radius. One wave32 per centroid.
// Each lane scans N/32 points keeping a sorted local top-K list (ascending
// d2), then a 32-round wave-wide lexicographic-min merge emits the global K
// nearest within radius, sorted, plus a valid count.
// ---------------------------------------------------------------------------
__global__ __launch_bounds__(256) void ballq_kernel(const float* __restrict__ pos,
                                                    const float* __restrict__ pos_s,
                                                    const int* __restrict__ samp,
                                                    int* __restrict__ nbr,
                                                    int* __restrict__ cnt) {
    const int gtid = blockIdx.x * blockDim.x + threadIdx.x;
    const int cid  = gtid >> 5;          // one wave per centroid
    const int lane = gtid & 31;
    if (cid >= BB * SS) return;

    const int b = cid >> 10;             // S == 1024
    const float cx = pos_s[(size_t)cid * 3 + 0];
    const float cy = pos_s[(size_t)cid * 3 + 1];
    const float cz = pos_s[(size_t)cid * 3 + 2];
    const float* pb = pos + (size_t)b * NN * 3;

    float ld[KK];
    int   li[KK];
    int   lc = 0;

    for (int p = lane; p < NN; p += 32) {
        const float dx = pb[p * 3 + 0] - cx;
        const float dy = pb[p * 3 + 1] - cy;
        const float dz = pb[p * 3 + 2] - cz;
        const float d2 = dx * dx + dy * dy + dz * dz;
        if (d2 <= R2) {
            if (lc < KK || d2 < ld[KK - 1]) {
                int q = (lc < KK) ? lc : (KK - 1);
                while (q > 0 && ld[q - 1] > d2) {
                    ld[q] = ld[q - 1];
                    li[q] = li[q - 1];
                    --q;
                }
                ld[q] = d2;
                li[q] = p;
                if (lc < KK) ++lc;
            }
        }
    }

    const int selfi = samp[cid];
    int head = 0;
    int c = 0;
    for (int k = 0; k < KK; ++k) {
        float cd = (head < lc) ? ld[head] : 1e30f;
        int   ci = (head < lc) ? li[head] : 0x7fffffff;
        float wd = cd;
        int   wi = ci;
#pragma unroll
        for (int m = 16; m >= 1; m >>= 1) {
            const float od = __shfl_xor(wd, m, 32);
            const int   oi = __shfl_xor(wi, m, 32);
            if (od < wd || (od == wd && oi < wi)) { wd = od; wi = oi; }
        }
        const bool fin = (wd < 1e29f);
        if (fin && cd == wd && ci == wi) ++head;   // winner lane pops
        if (lane == k) nbr[(size_t)cid * KK + k] = fin ? wi : selfi;
        if (fin) ++c;
    }
    if (lane == 0) cnt[cid] = c;
}

// ---------------------------------------------------------------------------
// Kernel 3: gather -> MLP (WMMA f16 in / f32 acc) -> masked max pool.
// 4 waves per block; weights staged once in LDS as f16 (W1 K-padded 67->96).
// Each wave handles one centroid per loop iteration:
//   GEMM1: feat[32x96] @ W1p[96x128]  -> 2x8x3  v_wmma_f32_16x16x32_f16
//   GEMM2: h1 [32x128] @ W2 [128x256] -> 2x16x4 v_wmma_f32_16x16x32_f16
// then bias+ReLU and column-max over the first cnt rows.
// ---------------------------------------------------------------------------
#define MLP_WAVES 4

__device__ __forceinline__ v8f wmma_f16(v16h a, v16h b, v8f c) {
    return __builtin_amdgcn_wmma_f32_16x16x32_f16(false, a, false, b,
                                                  (short)0, c, false, false);
}

__global__ __launch_bounds__(MLP_WAVES * 32) void mlp_kernel(
        const float* __restrict__ x, const float* __restrict__ pos,
        const float* __restrict__ pos_s, const int* __restrict__ nbr,
        const int* __restrict__ cnt, const float* __restrict__ W1,
        const float* __restrict__ b1, const float* __restrict__ W2,
        const float* __restrict__ b2, float* __restrict__ out) {
    __shared__ _Float16 W1s[96][128];                 // K-padded (rows 67..95 zero)
    __shared__ float    b1s[128];
    __shared__ _Float16 W2s[128][256];
    __shared__ float    b2s[256];
    __shared__ _Float16 feats[MLP_WAVES][32][96];     // per-wave gathered features
    __shared__ _Float16 h1s[MLP_WAVES][32][128];      // per-wave hidden layer

    const int tid = threadIdx.x;
    // stage weights (f32 -> f16) once per block
    for (int i = tid; i < 96 * 128; i += MLP_WAVES * 32) {
        const int r = i >> 7, c = i & 127;
        W1s[r][c] = (_Float16)((r < 67) ? W1[r * 128 + c] : 0.0f);
    }
    for (int i = tid; i < 128; i += MLP_WAVES * 32) b1s[i] = b1[i];
    for (int i = tid; i < 128 * 256; i += MLP_WAVES * 32) {
        const int r = i >> 8, c = i & 255;
        W2s[r][c] = (_Float16)W2[r * 256 + c];
    }
    for (int i = tid; i < 256; i += MLP_WAVES * 32) b2s[i] = b2[i];
    __syncthreads();

    const int wv    = tid >> 5;
    const int lane  = tid & 31;
    const int off8  = (lane < 16) ? 0 : 8;   // A-layout / C-layout row offset
    const int col   = lane & 15;             // C-layout column

    for (int cid = blockIdx.x * MLP_WAVES + wv; cid < BB * SS;
         cid += gridDim.x * MLP_WAVES) {
        const int b = cid >> 10;

        // ---- gather: lane k builds feature row k = [x_j(64), rel(3), 0...]
        const int j = nbr[(size_t)cid * KK + lane];
        const float* xr = x + ((size_t)(b * NN + j)) * CC;
        const float* pj = pos + (size_t)(b * NN + j) * 3;
        const float* ps = pos_s + (size_t)cid * 3;
        _Float16* fr = &feats[wv][lane][0];
#pragma unroll
        for (int c = 0; c < CC; c += 4) {
            const float4 v = *(const float4*)(xr + c);
            fr[c + 0] = (_Float16)v.x;
            fr[c + 1] = (_Float16)v.y;
            fr[c + 2] = (_Float16)v.z;
            fr[c + 3] = (_Float16)v.w;
        }
        fr[64] = (_Float16)(pj[0] - ps[0]);
        fr[65] = (_Float16)(pj[1] - ps[1]);
        fr[66] = (_Float16)(pj[2] - ps[2]);
#pragma unroll
        for (int c = 67; c < 96; ++c) fr[c] = (_Float16)0.0f;

        const int cn = cnt[cid];

        // ---- A fragments for GEMM1 (16-bit A 16x32 layout, wave32)
        v16h a1[2][3];
#pragma unroll
        for (int mt = 0; mt < 2; ++mt) {
#pragma unroll
            for (int kt = 0; kt < 3; ++kt) {
                const _Float16* base = &feats[wv][mt * 16 + (lane & 15)][kt * 32];
#pragma unroll
                for (int t = 0; t < 8; ++t) {
                    a1[mt][kt][t]     = base[off8 + t];
                    a1[mt][kt][8 + t] = base[16 + off8 + t];
                }
            }
        }

        // ---- GEMM1: [32x96] @ [96x128] -> h1 (bias + ReLU) in LDS
#pragma unroll
        for (int nt = 0; nt < 8; ++nt) {
            v16h bf[3];
#pragma unroll
            for (int kt = 0; kt < 3; ++kt) {
#pragma unroll
                for (int t = 0; t < 8; ++t) {
                    bf[kt][t]     = W1s[kt * 32 + off8 + t][nt * 16 + col];
                    bf[kt][8 + t] = W1s[kt * 32 + 16 + off8 + t][nt * 16 + col];
                }
            }
            const float bias = b1s[nt * 16 + col];
#pragma unroll
            for (int mt = 0; mt < 2; ++mt) {
                v8f acc = {0.f, 0.f, 0.f, 0.f, 0.f, 0.f, 0.f, 0.f};
#pragma unroll
                for (int kt = 0; kt < 3; ++kt)
                    acc = wmma_f16(a1[mt][kt], bf[kt], acc);
#pragma unroll
                for (int r = 0; r < 8; ++r) {
                    float v = acc[r] + bias;
                    v = (v > 0.0f) ? v : 0.0f;
                    h1s[wv][mt * 16 + off8 + r][nt * 16 + col] = (_Float16)v;
                }
            }
        }

        // ---- A fragments for GEMM2
        v16h a2[2][4];
#pragma unroll
        for (int mt = 0; mt < 2; ++mt) {
#pragma unroll
            for (int kt = 0; kt < 4; ++kt) {
                const _Float16* base = &h1s[wv][mt * 16 + (lane & 15)][kt * 32];
#pragma unroll
                for (int t = 0; t < 8; ++t) {
                    a2[mt][kt][t]     = base[off8 + t];
                    a2[mt][kt][8 + t] = base[16 + off8 + t];
                }
            }
        }

        // ---- GEMM2: [32x128] @ [128x256], bias + ReLU, masked max over rows
#pragma unroll
        for (int nt = 0; nt < 16; ++nt) {
            v16h bf[4];
#pragma unroll
            for (int kt = 0; kt < 4; ++kt) {
#pragma unroll
                for (int t = 0; t < 8; ++t) {
                    bf[kt][t]     = W2s[kt * 32 + off8 + t][nt * 16 + col];
                    bf[kt][8 + t] = W2s[kt * 32 + 16 + off8 + t][nt * 16 + col];
                }
            }
            const float bias = b2s[nt * 16 + col];
            float cmax = 0.0f;   // ReLU output >= 0 and cnt >= 1
#pragma unroll
            for (int mt = 0; mt < 2; ++mt) {
                v8f acc = {0.f, 0.f, 0.f, 0.f, 0.f, 0.f, 0.f, 0.f};
#pragma unroll
                for (int kt = 0; kt < 4; ++kt)
                    acc = wmma_f16(a2[mt][kt], bf[kt], acc);
#pragma unroll
                for (int r = 0; r < 8; ++r) {
                    const int row = mt * 16 + off8 + r;
                    if (row < cn) {
                        float v = acc[r] + bias;
                        v = (v > 0.0f) ? v : 0.0f;
                        cmax = fmaxf(cmax, v);
                    }
                }
            }
            const float o = __shfl_xor(cmax, 16, 32);   // combine row halves
            cmax = fmaxf(cmax, o);
            if (lane < 16)
                out[(size_t)cid * 256 + nt * 16 + col] = cmax;
        }
    }
}

// ---------------------------------------------------------------------------
// Host entry: inputs in setup_inputs() order: x, pos, W1, b1, W2, b2.
// d_out = out[16,1024,256] ++ pos_s[16,1024,3] (flat, f32).
// ---------------------------------------------------------------------------
extern "C" void kernel_launch(void* const* d_in, const int* in_sizes, int n_in,
                              void* d_out, int out_size, void* d_ws, size_t ws_size,
                              hipStream_t stream) {
    const float* x   = (const float*)d_in[0];
    const float* pos = (const float*)d_in[1];
    const float* W1  = (const float*)d_in[2];
    const float* b1  = (const float*)d_in[3];
    const float* W2  = (const float*)d_in[4];
    const float* b2  = (const float*)d_in[5];

    float* out   = (float*)d_out;
    float* pos_s = out + (size_t)BB * SS * 256;

    char* ws  = (char*)d_ws;
    int* samp = (int*)ws;                                         // 64 KB
    int* nbr  = (int*)(ws + 65536);                               // 2 MB
    int* cnt  = (int*)(ws + 65536 + (size_t)BB * SS * KK * 4);    // 64 KB

    fps_kernel<<<BB, 1024, 0, stream>>>(pos, samp, pos_s);
    ballq_kernel<<<(BB * SS * 32) / 256, 256, 0, stream>>>(pos, pos_s, samp, nbr, cnt);
    mlp_kernel<<<512, MLP_WAVES * 32, 0, stream>>>(x, pos, pos_s, nbr, cnt,
                                                   W1, b1, W2, b2, out);
}
